// Attention_5901285064931
// MI455X (gfx1250) — matrix operand order repaired and checked by reading, not compiled, not run on hardware
//
#include <hip/hip_runtime.h>
#include <hip/hip_bf16.h>

// ---------------------------------------------------------------------------
// Attention layer for MI455X (gfx1250): B=2, S=2048, E=1024, H=16, D=64
// bf16 WMMA everywhere (v_wmma_f32_16x16x32_bf16), f32 accumulation.
// Flash attention computed TRANSPOSED (S^T = K Q^T, O^T = V^T P^T): softmax
// stats lane-local (2 shuffles/tile), exp2-domain, mask-free interior tiles,
// K/V staged with global_load_async_to_lds_b128 (ASYNCcnt).
// ---------------------------------------------------------------------------

#define B_  2
#define S_  2048
#define E_  1024
#define H_  16
#define D_  64

typedef __bf16 bf16_t;
typedef __bf16  v16bf __attribute__((ext_vector_type(16)));
typedef __bf16  v8bf  __attribute__((ext_vector_type(8)));
typedef float   v8f   __attribute__((ext_vector_type(8)));

static __device__ __forceinline__ v8f wmma_bf16(v16bf a, v16bf b, v8f c) {
    return __builtin_amdgcn_wmma_f32_16x16x32_bf16(false, a, false, b,
                                                   (short)0, c, false, false);
}
static __device__ __forceinline__ v16bf cat16(v8bf a, v8bf b) {
    return __builtin_shufflevector(a, b, 0, 1, 2, 3, 4, 5, 6, 7,
                                         8, 9, 10, 11, 12, 13, 14, 15);
}
static __device__ __forceinline__ unsigned pack_bf16x2(float a, float b) {
    union { bf16_t h[2]; unsigned u; } p;
    p.h[0] = (bf16_t)a; p.h[1] = (bf16_t)b;
    return p.u;
}
static __device__ __forceinline__ float exp2_fast(float x) {
    return __builtin_amdgcn_exp2f(x);       // native v_exp_f32 (exp2)
}
// async 16B copy global -> LDS (per lane), tracked by ASYNCcnt
static __device__ __forceinline__ void async_copy_b128(void* lds, const void* gptr) {
    unsigned lofs = (unsigned)(__SIZE_TYPE__)lds;   // low 32 bits = LDS offset
    asm volatile("global_load_async_to_lds_b128 %0, %1, off"
                 :: "v"(lofs), "v"(gptr) : "memory");
}
static __device__ __forceinline__ void wait_async0() {
    asm volatile("s_wait_asynccnt 0" ::: "memory");
}

// ---------------------------------------------------------------------------
// Kernel 0a: f32 -> bf16 row-major convert (for x). 8 elems/thread.
// ---------------------------------------------------------------------------
__global__ __launch_bounds__(256)
void convert_bf16_kernel(const float* __restrict__ in, unsigned* __restrict__ out) {
    size_t i0 = ((size_t)blockIdx.x * 256 + threadIdx.x) * 8;
#pragma unroll
    for (int e = 0; e < 4; ++e) {
        float2 v = *(const float2*)&in[i0 + 2 * e];
        out[i0 / 2 + e] = pack_bf16x2(v.x, v.y);
    }
}

// ---------------------------------------------------------------------------
// Kernel 0b: f32 [K,N] -> bf16 [N,K] tiled transpose+convert (for weights).
// ---------------------------------------------------------------------------
__global__ __launch_bounds__(256)
void transpose_bf16_kernel(const float* __restrict__ in, unsigned* __restrict__ outT,
                           int K, int N) {
    __shared__ float t[64][33];
    const int tid = threadIdx.x;
    const int k0  = blockIdx.y * 64;
    const int n0  = blockIdx.x * 32;
#pragma unroll
    for (int i = 0; i < 8; ++i) {
        int idx = tid + i * 256;
        int kk = idx >> 5, nn = idx & 31;
        t[kk][nn] = in[(size_t)(k0 + kk) * N + n0 + nn];
    }
    __syncthreads();
    const int n  = tid >> 3;            // 0..31
    const int kb = (tid & 7) * 8;       // 0..56
#pragma unroll
    for (int e = 0; e < 4; ++e)
        outT[((size_t)(n0 + n) * K + k0 + kb) / 2 + e] =
            pack_bf16x2(t[kb + 2 * e][n], t[kb + 2 * e + 1][n]);
}

// ---------------------------------------------------------------------------
// Kernel 0c: bf16 [bh][s][d] -> bf16 [bh][d][s] transpose (for V).
// ---------------------------------------------------------------------------
__global__ __launch_bounds__(256)
void vtrans_kernel(const bf16_t* __restrict__ vb, bf16_t* __restrict__ vbT) {
    __shared__ bf16_t t[64][72];        // [s][d], 144B rows
    const int tid = threadIdx.x;
    const int bh  = blockIdx.y;
    const int s0  = blockIdx.x * 64;
    const bf16_t* src = vb + (size_t)bh * S_ * D_;
    bf16_t*       dst = vbT + (size_t)bh * D_ * S_;

    const int row = tid >> 2;           // s 0..63
    const int c0  = (tid & 3) * 16;     // d chunk
    *(v8bf*)&t[row][c0]     = *(const v8bf*)(src + (size_t)(s0 + row) * D_ + c0);
    *(v8bf*)&t[row][c0 + 8] = *(const v8bf*)(src + (size_t)(s0 + row) * D_ + c0 + 8);
    __syncthreads();

    const int d  = tid >> 2;            // 0..63
    const int sc = (tid & 3) * 16;      // s chunk
    v8bf o0, o1;
#pragma unroll
    for (int e = 0; e < 8; ++e) o0[e] = t[sc + e][d];
#pragma unroll
    for (int e = 0; e < 8; ++e) o1[e] = t[sc + 8 + e][d];
    *(v8bf*)(dst + (size_t)d * S_ + s0 + sc)     = o0;
    *(v8bf*)(dst + (size_t)d * S_ + s0 + sc + 8) = o1;
}

// ---------------------------------------------------------------------------
// Kernel 1/4: bf16 x bf16^T WMMA GEMM, f32 out.  C[M,N] = A[M,K] * BT[N,K]^T
// Block tile 128x64, 8 waves, each wave 32x32 (2x2 WMMA), K-step 32.
// ---------------------------------------------------------------------------
template <int N, int K>
__global__ __launch_bounds__(256)
void gemm_bb_kernel(const bf16_t* __restrict__ A, const bf16_t* __restrict__ BT,
                    float* __restrict__ C) {
    __shared__ bf16_t As[128][40];   // [m][k]
    __shared__ bf16_t Bs[64][40];    // [n][k]

    const int tid  = threadIdx.x;
    const int lane = tid & 31;
    const int wave = tid >> 5;
    const int wm   = wave & 3;
    const int wn   = wave >> 2;
    const int row0 = blockIdx.y * 128;
    const int col0 = blockIdx.x * 64;

    const int mlo  = lane & 15;
    const int hi   = lane >> 4;
    const int koff = hi * 8;

    const int ar  = tid >> 1;
    const int ac0 = (tid & 1) * 16;
    const int bn  = tid >> 2;
    const int bk0 = (tid & 3) * 8;

    const bf16_t* ap = A  + (size_t)(row0 + ar) * K + ac0;
    const bf16_t* bp = BT + (size_t)(col0 + bn) * K + bk0;

    v8f acc[2][2];
#pragma unroll
    for (int i = 0; i < 2; ++i)
#pragma unroll
        for (int j = 0; j < 2; ++j)
#pragma unroll
            for (int r = 0; r < 8; ++r) acc[i][j][r] = 0.0f;

    for (int k0 = 0; k0 < K; k0 += 32) {
        async_copy_b128(&As[ar][ac0],     ap + k0);
        async_copy_b128(&As[ar][ac0 + 8], ap + k0 + 8);
        async_copy_b128(&Bs[bn][bk0],     bp + k0);
        if (k0 + 32 < K) {
            __builtin_prefetch(ap + k0 + 32, 0, 3);
            __builtin_prefetch(bp + k0 + 32, 0, 3);
        }
        wait_async0();
        __syncthreads();

        v16bf bfr[2];
#pragma unroll
        for (int j = 0; j < 2; ++j) {
            int ncol = wn * 32 + j * 16 + mlo;
            int kb   = hi * 16;
            bfr[j] = cat16(*(const v8bf*)&Bs[ncol][kb],
                           *(const v8bf*)&Bs[ncol][kb + 8]);
        }
#pragma unroll
        for (int i = 0; i < 2; ++i) {
            int mrow = wm * 32 + i * 16 + mlo;
            v16bf af = cat16(*(const v8bf*)&As[mrow][hi * 16],
                             *(const v8bf*)&As[mrow][hi * 16 + 8]);
#pragma unroll
            for (int j = 0; j < 2; ++j)
                acc[i][j] = wmma_bf16(af, bfr[j], acc[i][j]);
        }
        __syncthreads();
    }

#pragma unroll
    for (int i = 0; i < 2; ++i)
#pragma unroll
        for (int j = 0; j < 2; ++j) {
            float* cp = C + (size_t)(row0 + wm * 32 + i * 16 + hi * 8) * N
                          + col0 + wn * 32 + j * 16 + mlo;
#pragma unroll
            for (int r = 0; r < 8; ++r)
                cp[(size_t)r * N] = acc[i][j][r];
        }
}

// ---------------------------------------------------------------------------
// Kernel 2: RoPE on q,k + repack q,k,v into bf16 [B,H,S,D].
// ---------------------------------------------------------------------------
__global__ __launch_bounds__(256)
void rope_pack_kernel(const float* __restrict__ qkv,
                      const float* __restrict__ cosT, const float* __restrict__ sinT,
                      unsigned* __restrict__ qb, unsigned* __restrict__ kb,
                      unsigned* __restrict__ vb) {
    int flat = blockIdx.x * 8 + (threadIdx.x >> 5);   // (b*S + s)*H + h
    int lane = threadIdx.x & 31;                       // pair index, D/2 = 32
    int h = flat % H_;
    int s = (flat / H_) % S_;
    int b = flat / (H_ * S_);

    const float* row = qkv + (size_t)(b * S_ + s) * (3 * E_);
    float c  = cosT[s * 32 + lane];
    float sn = sinT[s * 32 + lane];

    size_t ob = (((size_t)(b * H_ + h) * S_ + s) * D_) / 2 + lane;  // uint index

    float2 q = *(const float2*)&row[h * D_ + 2 * lane];
    qb[ob] = pack_bf16x2(q.x * c - q.y * sn, q.x * sn + q.y * c);

    float2 k = *(const float2*)&row[E_ + h * D_ + 2 * lane];
    kb[ob] = pack_bf16x2(k.x * c - k.y * sn, k.x * sn + k.y * c);

    float2 v = *(const float2*)&row[2 * E_ + h * D_ + 2 * lane];
    vb[ob] = pack_bf16x2(v.x, v.y);
}

// ---------------------------------------------------------------------------
// Kernel 3: causal flash attention, TRANSPOSED tiles.
// Grid: (S/128, B*H). Block: 256 threads / 8 waves; each wave = 16 queries.
// S^T = K(32k x 64d) @ Q^T(64d x 16q): 4 WMMAs; lane = one query column.
// O^T = V^T(64d x 32k) @ P^T(32k x 16q): 4 WMMAs. Softmax lane-local, exp2.
// ---------------------------------------------------------------------------
__global__ __launch_bounds__(256)
void flash_attn_kernel(const bf16_t* __restrict__ qb, const bf16_t* __restrict__ kb,
                       const bf16_t* __restrict__ vbT, bf16_t* __restrict__ out) {
    __shared__ bf16_t Ks[32][72];        // [key][d], 144B rows
    __shared__ bf16_t Vt[64][40];        // [d][key], 80B rows
    __shared__ bf16_t Ps[8][16][40];     // per-wave P [query][key]

    const int tid  = threadIdx.x;
    const int lane = tid & 31;
    const int wave = tid >> 5;
    const int mlo  = lane & 15;
    const int hi   = lane >> 4;
    const int koff = hi * 8;

    const int bh    = blockIdx.y;            // b*H + h
    const int b     = bh / H_;
    const int h     = bh % H_;
    const int qbase = blockIdx.x * 128 + wave * 16;

    const bf16_t* Qh  = qb  + (size_t)bh * S_ * D_;
    const bf16_t* Kh  = kb  + (size_t)bh * S_ * D_;
    const bf16_t* VhT = vbT + (size_t)bh * D_ * S_;   // [d][s]

    // staging coordinates (one async b128 each)
    const int skk = tid >> 3;            // K: key 0..31
    const int sd0 = (tid & 7) * 8;       // K: d base
    const int vdd = tid >> 2;            // V^T: d 0..63
    const int vk0 = (tid & 3) * 8;       // V^T: key base

    // Q^T B-fragments: lane = query mlo, elements = d (c*32 + hi*16 + e)
    v16bf qfb[2];
    {
        const bf16_t* qp = Qh + (size_t)(qbase + mlo) * D_;
#pragma unroll
        for (int c = 0; c < 2; ++c)
            qfb[c] = cat16(*(const v8bf*)(qp + c * 32 + hi * 16),
                           *(const v8bf*)(qp + c * 32 + hi * 16 + 8));
    }

    v8f o[4];
#pragma unroll
    for (int dt = 0; dt < 4; ++dt)
#pragma unroll
        for (int r = 0; r < 8; ++r) o[dt][r] = 0.0f;
    float m = -1e30f, l = 0.0f;

    const int qg  = qbase + mlo;             // this lane's query row
    const int nkt = blockIdx.x * 4 + 4;      // 32-key tiles covering causal range
    const float SCL2 = 0.125f * 1.44269504088896340736f;  // (1/sqrt(D))*log2(e)

    for (int kt = 0; kt < nkt; ++kt) {
        const int kb0 = kt * 32;
        async_copy_b128(&Ks[skk][sd0], Kh + (size_t)(kb0 + skk) * D_ + sd0);
        async_copy_b128(&Vt[vdd][vk0], VhT + (size_t)vdd * S_ + kb0 + vk0);
        wait_async0();
        __syncthreads();

        if (qbase + 15 >= kb0) {             // wave-uniform causal check
            // ---- S^T: two 16-key-group C-frags; lane = query, r = key ----
            v8f sfr[2];
#pragma unroll
            for (int g = 0; g < 2; ++g) {
                v8f s;
#pragma unroll
                for (int r = 0; r < 8; ++r) s[r] = 0.0f;
#pragma unroll
                for (int c = 0; c < 2; ++c) {
                    const bf16_t* kp = &Ks[g * 16 + mlo][c * 32];
                    v16bf af = cat16(*(const v8bf*)(kp + hi * 16),
                                     *(const v8bf*)(kp + hi * 16 + 8));
                    s = wmma_bf16(af, qfb[c], s);
                }
                sfr[g] = s;
            }
            // ---- lane-local stats; keys kb0 + g*16 + 8*hi + r (exp2 domain) ----
            float sv[2][8];
            float vmax = -1e30f;
            if (kb0 + 31 <= qbase) {         // interior tile: no masking needed
#pragma unroll
                for (int g = 0; g < 2; ++g)
#pragma unroll
                    for (int r = 0; r < 8; ++r) {
                        float s = sfr[g][r] * SCL2;
                        sv[g][r] = s;
                        vmax = fmaxf(vmax, s);
                    }
            } else {                         // diagonal tile: sentinel mask
#pragma unroll
                for (int g = 0; g < 2; ++g)
#pragma unroll
                    for (int r = 0; r < 8; ++r) {
                        int key = kb0 + g * 16 + 8 * hi + r;
                        float s = (key > qg) ? -1e30f : sfr[g][r] * SCL2;
                        sv[g][r] = s;
                        vmax = fmaxf(vmax, s);
                    }
            }
            vmax = fmaxf(vmax, __shfl_xor(vmax, 16, 32));   // other 16 keys
            float nm   = fmaxf(m, vmax);
            float corr = exp2_fast(m - nm);
            float sum  = 0.0f;
            v8bf pk[2];
#pragma unroll
            for (int g = 0; g < 2; ++g)
#pragma unroll
                for (int r = 0; r < 8; ++r) {
                    float p = exp2_fast(sv[g][r] - nm);   // masked -> underflow to 0
                    pk[g][r] = (bf16_t)p;
                    sum += p;
                }
            sum += __shfl_xor(sum, 16, 32);
            l = l * corr + sum;
            m = nm;
#pragma unroll
            for (int dt = 0; dt < 4; ++dt)
#pragma unroll
                for (int r = 0; r < 8; ++r) o[dt][r] *= corr;   // scalar rescale

            // ---- write P rows (8 contiguous keys per store) ----
            *(v8bf*)&Ps[wave][mlo][8 * hi]      = pk[0];
            *(v8bf*)&Ps[wave][mlo][16 + 8 * hi] = pk[1];

            // ---- O^T += V^T @ P^T ----
            v16bf pfb = cat16(*(const v8bf*)&Ps[wave][mlo][hi * 16],
                              *(const v8bf*)&Ps[wave][mlo][hi * 16 + 8]);
#pragma unroll
            for (int dt = 0; dt < 4; ++dt) {
                const bf16_t* vp = &Vt[dt * 16 + mlo][0];
                v16bf af = cat16(*(const v8bf*)(vp + hi * 16),
                                 *(const v8bf*)(vp + hi * 16 + 8));
                o[dt] = wmma_bf16(af, pfb, o[dt]);
            }
        }
        __syncthreads();
    }

    // ---- store: lane owns query qbase+mlo; 8 contiguous d per frag ----
    float invl = 1.0f / l;
    bf16_t* op = out + (size_t)(b * S_ + qbase + mlo) * E_ + h * D_ + 8 * hi;
#pragma unroll
    for (int dt = 0; dt < 4; ++dt) {
        v8bf t;
#pragma unroll
        for (int r = 0; r < 8; ++r) t[r] = (bf16_t)(o[dt][r] * invl);
        *(v8bf*)(op + dt * 16) = t;
    }
}

// ---------------------------------------------------------------------------
// Host launcher
// ---------------------------------------------------------------------------
extern "C" void kernel_launch(void* const* d_in, const int* in_sizes, int n_in,
                              void* d_out, int out_size, void* d_ws, size_t ws_size,
                              hipStream_t stream) {
    (void)in_sizes; (void)n_in; (void)out_size; (void)ws_size;

    const float* x     = (const float*)d_in[0];   // [B,S,E]
    const float* w_qkv = (const float*)d_in[1];   // [E,3E]
    const float* w_out = (const float*)d_in[2];   // [E,E]
    const float* fcos  = (const float*)d_in[3];   // [S,D/2]
    const float* fsin  = (const float*)d_in[4];   // [S,D/2]
    float*       outp  = (float*)d_out;           // [B,S,E]

    char* ws = (char*)d_ws;
    const size_t M   = (size_t)B_ * S_;           // 4096
    const size_t BHS = (size_t)B_ * H_ * S_ * D_; // per-tensor bf16 elems
    size_t off = 0;
    float*  qkv   = (float*)(ws + off);  off += M * (3 * E_) * sizeof(float);
    bf16_t* xb    = (bf16_t*)(ws + off); off += M * E_ * sizeof(bf16_t);
    bf16_t* wqkvT = (bf16_t*)(ws + off); off += (size_t)(3 * E_) * E_ * sizeof(bf16_t);
    bf16_t* woutT = (bf16_t*)(ws + off); off += (size_t)E_ * E_ * sizeof(bf16_t);
    bf16_t* qb    = (bf16_t*)(ws + off); off += BHS * sizeof(bf16_t);
    bf16_t* kb    = (bf16_t*)(ws + off); off += BHS * sizeof(bf16_t);
    bf16_t* vb    = (bf16_t*)(ws + off); off += BHS * sizeof(bf16_t);
    bf16_t* vbT   = (bf16_t*)(ws + off); off += BHS * sizeof(bf16_t);
    bf16_t* attb  = (bf16_t*)(ws + off); off += M * E_ * sizeof(bf16_t);

    // 0) one-time conversions (L2-resident afterwards)
    convert_bf16_kernel<<<(int)(M * E_ / 2048), 256, 0, stream>>>(x, (unsigned*)xb);
    {
        dim3 g1((3 * E_) / 32, E_ / 64);
        transpose_bf16_kernel<<<g1, 256, 0, stream>>>(w_qkv, (unsigned*)wqkvT, E_, 3 * E_);
        dim3 g2(E_ / 32, E_ / 64);
        transpose_bf16_kernel<<<g2, 256, 0, stream>>>(w_out, (unsigned*)woutT, E_, E_);
    }
    // 1) QKV projection: [4096,1024] @ [1024,3072]
    {
        dim3 grid((3 * E_) / 64, (int)M / 128);
        gemm_bb_kernel<3 * E_, E_><<<grid, 256, 0, stream>>>(xb, wqkvT, qkv);
    }
    // 2) RoPE + pack to bf16 [B,H,S,D]
    {
        int total = B_ * S_ * H_;
        rope_pack_kernel<<<total / 8, 256, 0, stream>>>(
            qkv, fcos, fsin, (unsigned*)qb, (unsigned*)kb, (unsigned*)vb);
    }
    // 2b) transpose V to [B,H,D,S]
    {
        dim3 grid(S_ / 64, B_ * H_);
        vtrans_kernel<<<grid, 256, 0, stream>>>(vb, vbT);
    }
    // 3) causal flash attention -> bf16 [B,S,E]
    {
        dim3 grid(S_ / 128, B_ * H_);
        flash_attn_kernel<<<grid, 256, 0, stream>>>(qb, kb, vbT, attb);
    }
    // 4) output projection: [4096,1024] @ [1024,1024]
    {
        dim3 grid(E_ / 64, (int)M / 128);
        gemm_bb_kernel<E_, E_><<<grid, 256, 0, stream>>>(attb, woutT, outp);
    }
}
